// STConvBlock_62577673503660
// MI455X (gfx1250) — matrix operand order repaired and checked by reading, not compiled, not run on hardware
//
#include <hip/hip_runtime.h>
#include <hip/hip_bf16.h>
#include <math.h>

typedef __attribute__((ext_vector_type(16))) _Float16 v16h;
typedef __attribute__((ext_vector_type(8)))  float    v8f;

#define B_  2
#define T_  12
#define T1_ 10
#define T2_ 8
#define N_  1024
#define C_  64
#define HK_ 6     // H * K1
#define R1_ 3     // R+1
#define MAXNZ 128 // fixed capacity per (k,row); expected nnz ~31

// ---- WMMA per-lane operand k order used here (contiguous k = 16*half + e, wave32) ----
__device__ __forceinline__ int a_kidx(int lane, int e) {
  return e + ((lane & 16) ? 16 : 0);
}
__device__ __forceinline__ void a_inv(int kk, int& inHigh, int& e) {
  inHigh = (kk >> 4) & 1;
  e = kk & 15;
}
// B (f16 32x16): lane L -> N=L&15, K = e + 16*(L>=16). Inverse: half=kk>>4, e=kk&15.

// ---- K0: conv weights -> f16 B-fragment layout [s(6)][a(8)][lane(32)][e(16)] ----
__global__ void wconv_kernel(const float* __restrict__ w1,
                             const float* __restrict__ w2,
                             _Float16* __restrict__ w1h,
                             _Float16* __restrict__ w2h) {
  int idx = blockIdx.x * 256 + threadIdx.x;
  if (idx >= 192 * 128) return;
  int kk = idx / 128, co = idx % 128;
  int kt = kk / 64, ci = kk % 64;
  int src = (co * 64 + ci) * 3 + kt;
  int s = kk >> 5, kkc = kk & 31;
  int laneF = (co & 15) + 16 * (kkc >> 4);
  int dst = ((s * 8 + (co >> 4)) * 32 + laneF) * 16 + (kkc & 15);
  w1h[dst] = (_Float16)w1[src];
  w2h[dst] = (_Float16)w2[src];
}

// ---- K0b: Wt -> f16 B-fragment layout [hk(6)][s(2)][a(4)][lane][e] ----
__global__ void wtfrag_kernel(const float* __restrict__ Wt, _Float16* __restrict__ wtf) {
  int idx = blockIdx.x * 256 + threadIdx.x;
  if (idx >= HK_ * C_ * C_) return;
  int hk = idx / (C_ * C_);
  int rem = idx - hk * (C_ * C_);
  int kk = rem / C_, col = rem % C_;
  int kkc = kk & 31;
  int laneF = (col & 15) + 16 * (kkc >> 4);
  int dst = (((hk * 2 + (kk >> 5)) * 4 + (col >> 4)) * 32 + laneF) * 16 + (kkc & 15);
  wtf[dst] = (_Float16)Wt[idx];
}

// ---- K1: mask bits: bit0=rel0!=0, bit1=rel1!=0, bit2=sup_k!=0 ----
__global__ void mask_kernel(const float* __restrict__ sup,
                            const float* __restrict__ rel,
                            unsigned char* __restrict__ mb) {
  int idx = blockIdx.x * 256 + threadIdx.x;
  if (idx >= 3 * N_ * N_) return;
  int k = idx / (N_ * N_);
  int ij = idx - k * (N_ * N_);
  unsigned char m = 0;
  if (rel[ij] != 0.f)                       m |= 1;
  if (rel[N_ * N_ + ij] != 0.f)             m |= 2;
  if (sup[(size_t)k * N_ * N_ + ij] != 0.f) m |= 4;
  mb[idx] = m;
}

// ---- K1b: fixed-capacity sparse row lists: ent = j | (bits<<10) ----
__global__ void csr_kernel(const unsigned char* __restrict__ mb,
                           unsigned short* __restrict__ ent,
                           unsigned short* __restrict__ cnt) {
  int rid = blockIdx.x * 256 + threadIdx.x;   // over 3*N rows (k,i)
  if (rid >= 3 * N_) return;
  const unsigned char* row = mb + (size_t)rid * N_;
  unsigned short* e = ent + (size_t)rid * MAXNZ;
  int c = 0;
  for (int j = 0; j < N_; j++) {
    unsigned char m = row[j];
    if (m && c < MAXNZ) e[c++] = (unsigned short)(j | ((int)m << 10));
  }
  cnt[rid] = (unsigned short)c;
}

// ---- K2/K6: temporal conv GEMM (16x128 tile, K=192) + GLU with residual ----
template <int Tin, int Tout>
__global__ void tconv_glu_kernel(const float* __restrict__ xin,
                                 const _Float16* __restrict__ whf,
                                 float* __restrict__ hout,
                                 _Float16* __restrict__ aout) {  // may be null
  int lane = threadIdx.x;
  int m0 = blockIdx.x * 16;
  int row = m0 + (lane & 15);
  int b  = row / (Tout * N_);
  int rm = row - b * (Tout * N_);
  int tp = rm / N_;
  int n  = rm - tp * N_;
  v8f c[8] = {};
  for (int s = 0; s < 6; s++) {
    v16h av;
#pragma unroll
    for (int e = 0; e < 16; e++) {
      int kk = 32 * s + a_kidx(lane, e);
      int kt = kk >> 6, ci = kk & 63;
      av[e] = (_Float16)xin[(((size_t)(b * Tin + tp + kt) * N_) + n) * C_ + ci];
    }
    v16h bv[8];
#pragma unroll
    for (int a = 0; a < 8; a++)
      bv[a] = *reinterpret_cast<const v16h*>(whf + ((s * 8 + a) * 32 + lane) * 16);
#pragma unroll
    for (int a = 0; a < 8; a++)
      c[a] = __builtin_amdgcn_wmma_f32_16x16x32_f16(
          false, av, false, bv[a], (short)0, c[a], false, false);
  }
  int rl = (lane & 16) ? 8 : 0;
#pragma unroll
  for (int v = 0; v < 8; v++) {
    int r = m0 + v + rl;
    int bb = r / (Tout * N_);
    int rr = r - bb * (Tout * N_);
    int tt = rr / N_;
    int nn = rr - tt * N_;
    int sl = bb * Tout + tt;
#pragma unroll
    for (int a = 0; a < 4; a++) {
      int col = 16 * a + (lane & 15);
      float res = xin[(((size_t)(bb * Tin + tt + 2) * N_) + nn) * C_ + col];
      float p = c[a][v] + res;
      float q = c[a + 4][v];
      float o = p * (1.f / (1.f + __expf(-q)));
      hout[(size_t)r * C_ + col] = o;
      if (aout) {  // A-fragment layout [sl][mtile(64)][st(2)][laneF][e]
        int kkc = col & 31, inH, e;
        a_inv(kkc, inH, e);
        size_t fi = ((((size_t)sl * 64 + (nn >> 4)) * 2 + (col >> 5)) * 32 +
                     ((nn & 15) + 16 * inH)) * 16 + e;
        aout[fi] = (_Float16)o;
      }
    }
  }
}

// ---- K3: wx = xs @ Wt[h,k]; store wx in B-fragment layout + al/ar ----
__global__ void wx_kernel(const _Float16* __restrict__ h1af,
                          const _Float16* __restrict__ wtf,
                          const float* __restrict__ Wl,
                          const float* __restrict__ Wr,
                          _Float16* __restrict__ wxh,
                          float* __restrict__ al,
                          float* __restrict__ ar) {
  __shared__ float wt_lds[16 * 64];
  int lane = threadIdx.x;
  int mtile = blockIdx.x, m0 = mtile * 16;
  int hk = blockIdx.y;
  int sl = blockIdx.z;
  v16h av[2], bv[2][4];
#pragma unroll
  for (int s = 0; s < 2; s++) {
    av[s] = *reinterpret_cast<const v16h*>(
        h1af + ((((size_t)sl * 64 + mtile) * 2 + s) * 32 + lane) * 16);
#pragma unroll
    for (int a = 0; a < 4; a++)
      bv[s][a] = *reinterpret_cast<const v16h*>(
          wtf + (((size_t)(hk * 2 + s) * 4 + a) * 32 + lane) * 16);
  }
  v8f c[4] = {};
#pragma unroll
  for (int s = 0; s < 2; s++)
#pragma unroll
    for (int a = 0; a < 4; a++)
      c[a] = __builtin_amdgcn_wmma_f32_16x16x32_f16(
          false, av[s], false, bv[s][a], (short)0, c[a], false, false);
  size_t wb = ((size_t)sl * HK_ + hk) * 32 * 4 * 32 * 16;
  int rl = (lane & 16) ? 8 : 0;
#pragma unroll
  for (int a = 0; a < 4; a++)
#pragma unroll
    for (int v = 0; v < 8; v++) {
      int r = m0 + v + rl;
      int col = 16 * a + (lane & 15);
      float val = c[a][v];
      wt_lds[(v + rl) * 64 + col] = val;
      int kkc = r & 31;
      int laneF = (lane & 15) + 16 * (kkc >> 4);
      size_t fi = wb + (((size_t)(r >> 5) * 4 + a) * 32 + laneF) * 16 + (kkc & 15);
      wxh[fi] = (_Float16)val;
    }
  __syncthreads();
  int r = lane & 15;
  const float* wlr = ((lane < 16) ? Wl : Wr) + (size_t)hk * R1_ * C_;
  float acc0 = 0.f, acc1 = 0.f, acc2 = 0.f;
  for (int cc = 0; cc < 64; cc++) {
    float xv = wt_lds[r * 64 + cc];
    acc0 += xv * wlr[0 * 64 + cc];
    acc1 += xv * wlr[1 * 64 + cc];
    acc2 += xv * wlr[2 * 64 + cc];
  }
  float* dst = (lane < 16) ? al : ar;
  size_t ab = (((size_t)sl * HK_ + hk) * N_ + (m0 + r)) * 3;
  dst[ab + 0] = acc0; dst[ab + 1] = acc1; dst[ab + 2] = acc2;
}

// ---- K4: sparse masked-softmax attention + P@WX (deferred 1/sum), ELU ----
__global__ void attn_kernel(const _Float16* __restrict__ wxh,
                            const float* __restrict__ al,
                            const float* __restrict__ ar,
                            const unsigned short* __restrict__ ent,
                            const unsigned short* __restrict__ cnt,
                            float* __restrict__ sheads) {
  __shared__ _Float16 scf[32 * 32 * 16];   // 32KB A-fragments [st][laneF][e]
  __shared__ float red[32];
  __shared__ float redinv[16];
  int lane = threadIdx.x;
  int m0 = blockIdx.x * 16;
  int h  = blockIdx.y;
  int sl = blockIdx.z;
  int irow = lane & 15;
  int half = lane >> 4;
  int i_g = m0 + irow;
  v8f c[4] = {};
  for (int k = 0; k < 3; k++) {
    int hk = h * 3 + k;
    size_t base = ((size_t)sl * HK_ + hk) * N_;
    float a0 = al[(base + i_g) * 3 + 0];
    float a1 = al[(base + i_g) * 3 + 1];
    float a2 = al[(base + i_g) * 3 + 2];
    const float* arp = ar + base * 3;
    const unsigned short* erow = ent + ((size_t)k * N_ + i_g) * MAXNZ;
    int nz = cnt[k * N_ + i_g];
    // zero the fragment buffer (zeros == masked == exp contribution 0)
    v16h zz = {};
#pragma unroll
    for (int st = 0; st < 32; st++)
      *reinterpret_cast<v16h*>(&scf[(st * 32 + lane) * 16]) = zz;
    __syncthreads();
    // pass 1 over sparse entries: row max (score==0 means masked -> skip)
    float mx = -__builtin_inff();
#pragma unroll 1
    for (int t = half; t < nz; t += 2) {
      unsigned short v = erow[t];
      int j = v & 1023, bits = v >> 10;
      float s = 0.f;
      if (bits & 1) s += a0 + arp[j * 3 + 0];
      if (bits & 2) s += a1 + arp[j * 3 + 1];
      if (bits & 4) s += a2 + arp[j * 3 + 2];
      if (s != 0.f) mx = fmaxf(mx, s);
    }
    red[lane] = mx; __syncthreads();
    mx = fmaxf(red[irow], red[irow + 16]); __syncthreads();
    // pass 2: recompute, exp, scatter into fragment slots, row sum
    float sum = 0.f;
#pragma unroll 1
    for (int t = half; t < nz; t += 2) {
      unsigned short v = erow[t];
      int j = v & 1023, bits = v >> 10;
      float s = 0.f;
      if (bits & 1) s += a0 + arp[j * 3 + 0];
      if (bits & 2) s += a1 + arp[j * 3 + 1];
      if (bits & 4) s += a2 + arp[j * 3 + 2];
      if (s != 0.f) {
        float ev = __expf(s - mx);
        sum += ev;
        int kkc = j & 31, inH, e;
        a_inv(kkc, inH, e);
        scf[(((j >> 5) * 32) + irow + 16 * inH) * 16 + e] = (_Float16)ev;
      }
    }
    red[lane] = sum; __syncthreads();
    sum = red[irow] + red[irow + 16];
    if (lane < 16) redinv[irow] = (sum > 0.f) ? 1.f / sum : 0.f;  // NaN->0
    __syncthreads();
    // WMMA: d = exp(P) @ WX with ping-pong double buffering; c += inv[row]*d
    const _Float16* wxb = wxh + ((size_t)sl * HK_ + hk) * 32 * 4 * 32 * 16;
    v8f d[4] = {};
    v16h avb[2], bvb[2][4];
    avb[0] = *reinterpret_cast<const v16h*>(&scf[(0 * 32 + lane) * 16]);
#pragma unroll
    for (int a = 0; a < 4; a++)
      bvb[0][a] = *reinterpret_cast<const v16h*>(wxb + ((size_t)a * 32 + lane) * 16);
#pragma unroll 2
    for (int st = 0; st < 32; st++) {
      int cur = st & 1, nxt = cur ^ 1;
      if (st < 31) {
        avb[nxt] = *reinterpret_cast<const v16h*>(&scf[((st + 1) * 32 + lane) * 16]);
#pragma unroll
        for (int a = 0; a < 4; a++)
          bvb[nxt][a] = *reinterpret_cast<const v16h*>(
              wxb + (((size_t)(st + 1) * 4 + a) * 32 + lane) * 16);
      }
#pragma unroll
      for (int a = 0; a < 4; a++)
        d[a] = __builtin_amdgcn_wmma_f32_16x16x32_f16(
            false, avb[cur], false, bvb[cur][a], (short)0, d[a], false, false);
    }
    int rl = (lane & 16) ? 8 : 0;
#pragma unroll
    for (int a = 0; a < 4; a++)
#pragma unroll
      for (int v = 0; v < 8; v++)
        c[a][v] += redinv[v + rl] * d[a][v];
    __syncthreads();
  }
  int rl = (lane & 16) ? 8 : 0;
  size_t ob = ((size_t)sl * 2 + h) * N_ * C_;
#pragma unroll
  for (int a = 0; a < 4; a++)
#pragma unroll
    for (int v = 0; v < 8; v++) {
      int r = m0 + v + rl, col = 16 * a + (lane & 15);
      float x = c[a][v];
      sheads[ob + (size_t)r * C_ + col] = (x > 0.f) ? x : (__expf(x) - 1.f);
    }
}

// ---- K5: average the two heads ----
__global__ void avg_heads_kernel(const float* __restrict__ sheads,
                                 float* __restrict__ sout) {
  int idx = blockIdx.x * 256 + threadIdx.x;
  if (idx >= B_ * T1_ * N_ * C_) return;
  int sl = idx / (N_ * C_);
  int rc = idx - sl * (N_ * C_);
  float a = sheads[((size_t)sl * 2 + 0) * N_ * C_ + rc];
  float b = sheads[((size_t)sl * 2 + 1) * N_ * C_ + rc];
  sout[idx] = 0.5f * (a + b);
}

// ---- K7: layer norm over (N,C) per (b,t) ----
__global__ void norm_kernel(const float* __restrict__ h2,
                            const float* __restrict__ gamma,
                            const float* __restrict__ beta,
                            float* __restrict__ out) {
  __shared__ float rs[256], rq[256];
  int bt = blockIdx.x;
  const float* src = h2 + (size_t)bt * N_ * C_;
  float s = 0.f, q = 0.f;
  for (int i = threadIdx.x; i < N_ * C_; i += 256) {
    float v = src[i]; s += v; q += v * v;
  }
  rs[threadIdx.x] = s; rq[threadIdx.x] = q; __syncthreads();
  for (int o = 128; o > 0; o >>= 1) {
    if ((int)threadIdx.x < o) {
      rs[threadIdx.x] += rs[threadIdx.x + o];
      rq[threadIdx.x] += rq[threadIdx.x + o];
    }
    __syncthreads();
  }
  float mu = rs[0] / (float)(N_ * C_);
  float var = rq[0] / (float)(N_ * C_) - mu * mu;
  float rn = rsqrtf(var + 1e-6f);
  for (int i = threadIdx.x; i < N_ * C_; i += 256)
    out[(size_t)bt * N_ * C_ + i] = (src[i] - mu) * rn * gamma[i] + beta[i];
}

extern "C" void kernel_launch(void* const* d_in, const int* in_sizes, int n_in,
                              void* d_out, int out_size, void* d_ws, size_t ws_size,
                              hipStream_t stream) {
  const float* x    = (const float*)d_in[0];
  const float* sup  = (const float*)d_in[1];
  const float* rel  = (const float*)d_in[2];
  const float* w_t1 = (const float*)d_in[3];
  const float* Wt   = (const float*)d_in[4];
  const float* Wl   = (const float*)d_in[5];
  const float* Wr   = (const float*)d_in[6];
  const float* w_t2 = (const float*)d_in[7];
  const float* gamma = (const float*)d_in[8];
  const float* beta  = (const float*)d_in[9];
  float* out = (float*)d_out;
  (void)in_sizes; (void)n_in; (void)out_size; (void)ws_size;

  char* ws = (char*)d_ws;
  size_t off = 0;
  auto alloc = [&](size_t bytes) -> void* {
    void* p = ws + off;
    off += (bytes + 255) & ~(size_t)255;
    return p;
  };
  float*          h1     = (float*)alloc((size_t)B_ * T1_ * N_ * C_ * 4);
  _Float16*       h1af   = (_Float16*)alloc((size_t)20 * N_ * C_ * 2);
  _Float16*       wxh    = (_Float16*)alloc((size_t)20 * HK_ * N_ * C_ * 2);
  float*          al     = (float*)alloc((size_t)20 * HK_ * N_ * 3 * 4);
  float*          ar     = (float*)alloc((size_t)20 * HK_ * N_ * 3 * 4);
  unsigned char*  maskb  = (unsigned char*)alloc((size_t)3 * N_ * N_);
  unsigned short* ent    = (unsigned short*)alloc((size_t)3 * N_ * MAXNZ * 2);
  unsigned short* cnt    = (unsigned short*)alloc((size_t)3 * N_ * 2);
  float*          sheads = (float*)alloc((size_t)20 * 2 * N_ * C_ * 4);
  float*          savg   = (float*)alloc((size_t)B_ * T1_ * N_ * C_ * 4);
  float*          h2     = (float*)alloc((size_t)B_ * T2_ * N_ * C_ * 4);
  _Float16*       w1h    = (_Float16*)alloc((size_t)192 * 128 * 2);
  _Float16*       w2h    = (_Float16*)alloc((size_t)192 * 128 * 2);
  _Float16*       wtf    = (_Float16*)alloc((size_t)HK_ * C_ * C_ * 2);

  wconv_kernel<<<96, 256, 0, stream>>>(w_t1, w_t2, w1h, w2h);
  wtfrag_kernel<<<(HK_ * C_ * C_) / 256, 256, 0, stream>>>(Wt, wtf);
  mask_kernel<<<(3 * N_ * N_) / 256, 256, 0, stream>>>(sup, rel, maskb);
  csr_kernel<<<(3 * N_ + 255) / 256, 256, 0, stream>>>(maskb, ent, cnt);
  tconv_glu_kernel<T_, T1_><<<(B_ * T1_ * N_) / 16, 32, 0, stream>>>(x, w1h, h1, h1af);
  wx_kernel<<<dim3(N_ / 16, HK_, 20), 32, 0, stream>>>(h1af, wtf, Wl, Wr, wxh, al, ar);
  attn_kernel<<<dim3(N_ / 16, 2, 20), 32, 0, stream>>>(wxh, al, ar, ent, cnt, sheads);
  avg_heads_kernel<<<(B_ * T1_ * N_ * C_) / 256, 256, 0, stream>>>(sheads, savg);
  tconv_glu_kernel<T1_, T2_><<<(B_ * T2_ * N_) / 16, 32, 0, stream>>>(savg, w2h, h2, nullptr);
  norm_kernel<<<B_ * T2_, 256, 0, stream>>>(h2, gamma, beta, out);
}
